// NTM_9423158247925
// MI455X (gfx1250) — matrix operand order repaired and checked by reading, not compiled, not run on hardware
//
#include <hip/hip_runtime.h>
#include <hip/hip_bf16.h>
#include <math.h>

// NTM forward for MI455X (gfx1250, wave32).
// Bandwidth plan: 8 sequential sweeps over the 256MB memory tensor dominate
// (~5GB traffic -> ~220us at 23.3 TB/s). Those passes stay f32 / coalesced.
// GEMMs (~2 GFLOP) run through v_wmma_f32_16x16x32_f16 with f32 accumulate.

#define Bdim 256
#define Nmem 4096
#define Mmem 64
#define Cdim 512
#define Hh   4
#define INd  64
#define OUTd 64
#define KXH  (INd + Hh*Mmem + Cdim)   // 832: [x_cat | h0]
#define G4C  (4*Cdim)                 // 2048
#define NPROJ (2*Hh*(Mmem+6) + Hh*2*Mmem)  // 560 + 512 = 1072
#define KST  (Cdim + Hh*Mmem)         // 768

typedef __attribute__((ext_vector_type(16))) _Float16 v16h;
typedef __attribute__((ext_vector_type(8)))  float    v8f;

__device__ __forceinline__ float sigm(float x) { return 1.f / (1.f + expf(-x)); }
__device__ __forceinline__ float softplus(float x) { return (x > 20.f) ? x : log1pf(expf(x)); }

// ---------------------------------------------------------------------------
// Pack kernels (f32 -> f16 staging for WMMA operands)
// ---------------------------------------------------------------------------
__global__ void ntm_pack_xh(const float* __restrict__ in_data,
                            const float* __restrict__ prev_reads,
                            const float* __restrict__ h0,
                            _Float16* __restrict__ xh16) {
  int i = blockIdx.x * 256 + threadIdx.x;
  if (i >= Bdim * KXH) return;
  int b = i / KXH, k = i % KXH;
  float v;
  if (k < INd)                 v = in_data[b * INd + k];
  else if (k < INd + Hh*Mmem)  { int kk = k - INd; v = prev_reads[((kk/Mmem)*Bdim + b)*Mmem + (kk%Mmem)]; }
  else                         v = h0[b * Cdim + (k - INd - Hh*Mmem)];
  xh16[i] = (_Float16)v;
}

__global__ void ntm_pack_w1(const float* __restrict__ W_ih,
                            const float* __restrict__ W_hh,
                            _Float16* __restrict__ w16) {
  int i = blockIdx.x * 256 + threadIdx.x;
  if (i >= G4C * KXH) return;
  int j = i / KXH, k = i % KXH;
  float v = (k < INd + Hh*Mmem) ? W_ih[j*(INd+Hh*Mmem) + k] : W_hh[j*Cdim + (k - INd - Hh*Mmem)];
  w16[i] = (_Float16)v;
}

__global__ void ntm_pack_half(const float* __restrict__ src, _Float16* __restrict__ dst, int n) {
  int i = blockIdx.x * 256 + threadIdx.x;
  if (i < n) dst[i] = (_Float16)src[i];
}

__global__ void ntm_pack_wp(const float* __restrict__ W_addr,
                            const float* __restrict__ W_ea,
                            _Float16* __restrict__ w16) {
  int i = blockIdx.x * 256 + threadIdx.x;
  if (i >= NPROJ * Cdim) return;
  int j = i / Cdim, k = i % Cdim;
  float v = (j < 2*Hh*(Mmem+6)) ? W_addr[j*Cdim + k] : W_ea[(j - 2*Hh*(Mmem+6))*Cdim + k];
  w16[i] = (_Float16)v;
}

__global__ void ntm_pack_biasp(const float* __restrict__ b_addr,
                               const float* __restrict__ b_ea,
                               float* __restrict__ bp) {
  int j = blockIdx.x * 256 + threadIdx.x;
  if (j >= NPROJ) return;
  bp[j] = (j < 2*Hh*(Mmem+6)) ? b_addr[j] : b_ea[j - 2*Hh*(Mmem+6)];
}

__global__ void ntm_pack_state(const float* __restrict__ h_state,
                               const float* __restrict__ reads,
                               _Float16* __restrict__ st16) {
  int i = blockIdx.x * 256 + threadIdx.x;
  if (i >= Bdim * KST) return;
  int b = i / KST, k = i % KST;
  float v;
  if (k < Cdim) v = h_state[b*Cdim + k];
  else { int kk = k - Cdim; v = reads[((kk/Mmem)*Bdim + b)*Mmem + (kk%Mmem)]; }
  st16[i] = (_Float16)v;
}

// ---------------------------------------------------------------------------
// WMMA GEMM: out[M,N] = A[M,K](f16,row-major) * Wt[N,K]^T(f16,row-major)
// one wave per 16x16 tile; K multiple of 32. mode: 0=raw 1=+bias 2=+bias+sigmoid
// A and B both use contiguous k order, k = 16*half + e (wave32).
// ---------------------------------------------------------------------------
__launch_bounds__(32)
__global__ void ntm_gemm16(const _Float16* __restrict__ A,
                           const _Float16* __restrict__ Wt,
                           float* __restrict__ Cout,
                           const float* __restrict__ bias,
                           int Mdim, int Ndim, int Kdim, int mode) {
  int lane = threadIdx.x;
  int row0 = blockIdx.x * 16, col0 = blockIdx.y * 16;
  int hsel = lane >> 4;        // lane half: 0 or 1
  int l15  = lane & 15;
  v8f acc = {};
  int arow = row0 + l15;
  int bcol = col0 + l15;
  for (int k0 = 0; k0 < Kdim; k0 += 32) {
    v16h a, bf;
    // A 16x32: lanes 0-15 hold K k0..k0+15; lanes 16-31 hold K+16
    const _Float16* Ap = A + (size_t)arow * Kdim + k0 + hsel * 16;
#pragma unroll
    for (int e = 0; e < 16; ++e) a[e] = Ap[e];
    // B 32x16 (= Wt^T): lanes 0-15 hold K k0..k0+15 of column bcol; lanes 16-31 hold K+16
    const _Float16* Wp = Wt + (size_t)bcol * Kdim + k0 + hsel * 16;
#pragma unroll
    for (int e = 0; e < 16; ++e) bf[e] = Wp[e];
    acc = __builtin_amdgcn_wmma_f32_16x16x32_f16(false, a, false, bf,
                                                 (short)0, acc, false, false);
  }
  float bv = bias ? bias[bcol] : 0.f;
#pragma unroll
  for (int r = 0; r < 8; ++r) {
    int row = row0 + hsel * 8 + r;     // C/D: VGPR r -> M=r (lanes 0-15) / M=r+8
    float v = acc[r] + bv;
    if (mode == 2) v = sigm(v);
    Cout[(size_t)row * Ndim + bcol] = v;
  }
}

// ---------------------------------------------------------------------------
// LSTM elementwise
// ---------------------------------------------------------------------------
__global__ void ntm_lstm(const float* __restrict__ gates,
                         const float* __restrict__ b_ih, const float* __restrict__ b_hh,
                         const float* __restrict__ c0,
                         float* __restrict__ c_state, float* __restrict__ h_state) {
  int i = blockIdx.x * 256 + threadIdx.x;
  if (i >= Bdim * Cdim) return;
  int b = i / Cdim, cc = i % Cdim;
  const float* g = gates + (size_t)b * G4C;
  float ig = g[cc]          + b_ih[cc]          + b_hh[cc];
  float fg = g[Cdim + cc]   + b_ih[Cdim + cc]   + b_hh[Cdim + cc];
  float gg = g[2*Cdim + cc] + b_ih[2*Cdim + cc] + b_hh[2*Cdim + cc];
  float og = g[3*Cdim + cc] + b_ih[3*Cdim + cc] + b_hh[3*Cdim + cc];
  float c = sigm(fg) * c0[i] + sigm(ig) * tanhf(gg);
  c_state[i] = c;
  h_state[i] = sigm(og) * tanhf(c);
}

// ---------------------------------------------------------------------------
// One NTM head (launched 8x, stream-ordered). One workgroup per batch row b:
// full addressing pipeline (cosine scores -> softmax -> interpolate -> shift
// -> sharpen -> normalize) in LDS, then read reduction or erase/add write.
// ---------------------------------------------------------------------------
__launch_bounds__(256)
__global__ void ntm_head(int hi,
                         const float* __restrict__ mem_in,
                         float* __restrict__ mem_out,
                         const float* __restrict__ proj,
                         const float* __restrict__ prev_w,
                         float* __restrict__ reads) {
  __shared__ float kv[Mmem];
  __shared__ float sc[Nmem];
  __shared__ float wg[Nmem];
  __shared__ float part[256];
  __shared__ float scal[1];

  int b = blockIdx.x, tid = threadIdx.x;
  const float* p = proj + (size_t)b * NPROJ + hi * (Mmem + 6);

  if (tid < Mmem) { float t = tanhf(p[tid]); kv[tid] = t; part[tid] = t * t; }
  __syncthreads();
  if (tid == 0) { float s = 0.f; for (int m = 0; m < Mmem; ++m) s += part[m]; scal[0] = sqrtf(s); }
  __syncthreads();
  float knorm = scal[0];
  float beta  = softplus(p[Mmem]);
  float gate  = sigm(p[Mmem + 1]);
  float s0p = p[Mmem + 2], s1p = p[Mmem + 3], s2p = p[Mmem + 4];
  float smx = fmaxf(s0p, fmaxf(s1p, s2p));
  float e0 = expf(s0p - smx), e1 = expf(s1p - smx), e2 = expf(s2p - smx);
  float einv = 1.f / (e0 + e1 + e2);
  float sh0 = e0 * einv, sh1 = e1 * einv, sh2 = e2 * einv;
  float gamma = 1.f + softplus(p[Mmem + 5]);

  const float* mbase = mem_in + (size_t)b * Nmem * Mmem;

  // -- phase 1: cosine-sim scores (coalesced: 16 lanes per memory row) --
  int q = tid & 15, rsub = tid >> 4;      // 16 rows per block pass
  for (int n0 = 0; n0 < Nmem; n0 += 16) {
    int n = n0 + rsub;
    float4 v = ((const float4*)(mbase + (size_t)n * Mmem))[q];
    float dot = v.x*kv[4*q] + v.y*kv[4*q+1] + v.z*kv[4*q+2] + v.w*kv[4*q+3];
    float nsq = v.x*v.x + v.y*v.y + v.z*v.z + v.w*v.w;
#pragma unroll
    for (int off = 8; off; off >>= 1) { dot += __shfl_xor(dot, off); nsq += __shfl_xor(nsq, off); }
    if (q == 0) sc[n] = beta * dot / (sqrtf(nsq) * knorm + 1e-8f);
  }
  __syncthreads();

  // -- phase 2: softmax over N --
  float lm = -1e30f;
  for (int n = tid; n < Nmem; n += 256) lm = fmaxf(lm, sc[n]);
  part[tid] = lm; __syncthreads();
  for (int s = 128; s > 0; s >>= 1) { if (tid < s) part[tid] = fmaxf(part[tid], part[tid + s]); __syncthreads(); }
  float mx = part[0]; __syncthreads();
  float ls = 0.f;
  for (int n = tid; n < Nmem; n += 256) { float e = expf(sc[n] - mx); sc[n] = e; ls += e; }
  part[tid] = ls; __syncthreads();
  for (int s = 128; s > 0; s >>= 1) { if (tid < s) part[tid] += part[tid + s]; __syncthreads(); }
  float Zinv = 1.f / part[0]; __syncthreads();

  // -- phase 3: interpolate, circular shift, sharpen, normalize --
  const float* pw = prev_w + ((size_t)hi * Bdim + b) * Nmem;
  for (int n = tid; n < Nmem; n += 256)
    wg[n] = gate * (sc[n] * Zinv) + (1.f - gate) * pw[n];
  __syncthreads();
  float lS = 0.f;
  for (int n = tid; n < Nmem; n += 256) {
    float ws = sh0 * wg[(n + Nmem - 1) & (Nmem - 1)] + sh1 * wg[n] + sh2 * wg[(n + 1) & (Nmem - 1)];
    float wp = __powf(ws, gamma);        // ws >= 0
    sc[n] = wp; lS += wp;
  }
  part[tid] = lS; __syncthreads();
  for (int s = 128; s > 0; s >>= 1) { if (tid < s) part[tid] += part[tid + s]; __syncthreads(); }
  float invS = 1.f / (part[0] + 1e-8f); __syncthreads();
  for (int n = tid; n < Nmem; n += 256) sc[n] *= invS;
  __syncthreads();

  // -- phase 4: read reduction or erase/add write (coalesced over m) --
  int m = tid & 63, stripe = tid >> 6;
  if ((hi & 1) == 0) {                    // read head
    float acc = 0.f;
    for (int n = stripe; n < Nmem; n += 4) {
      __builtin_prefetch(mbase + (size_t)(n + 64) * Mmem + m, 0, 0);
      acc += sc[n] * mbase[(size_t)n * Mmem + m];
    }
    part[tid] = acc; __syncthreads();
    if (tid < 64)
      reads[((size_t)(hi >> 1) * Bdim + b) * Mmem + m] =
          part[tid] + part[64 + tid] + part[128 + tid] + part[192 + tid];
  } else {                                // write head: erase then add
    const float* ea = proj + (size_t)b * NPROJ + 2*Hh*(Mmem + 6) + (hi >> 1) * (2 * Mmem);
    float em = sigm(ea[m]);
    float am = tanhf(ea[Mmem + m]);
    float* obase = mem_out + (size_t)b * Nmem * Mmem;
    for (int n = stripe; n < Nmem; n += 4) {
      size_t idx = (size_t)n * Mmem + m;
      float v = mbase[idx];
      float w = sc[n];
      obase[idx] = v * (1.f - w * em) + w * am;
    }
  }
}

// ---------------------------------------------------------------------------
extern "C" void kernel_launch(void* const* d_in, const int* in_sizes, int n_in,
                              void* d_out, int out_size, void* d_ws, size_t ws_size,
                              hipStream_t stream) {
  const float* in_data    = (const float*)d_in[0];
  const float* memory     = (const float*)d_in[1];
  const float* h0         = (const float*)d_in[2];
  const float* c0         = (const float*)d_in[3];
  const float* prev_w     = (const float*)d_in[4];
  const float* prev_reads = (const float*)d_in[5];
  const float* W_ih       = (const float*)d_in[6];
  const float* b_ih       = (const float*)d_in[7];
  const float* W_hh       = (const float*)d_in[8];
  const float* b_hh       = (const float*)d_in[9];
  const float* W_out      = (const float*)d_in[10];
  const float* b_out      = (const float*)d_in[11];
  const float* W_addr     = (const float*)d_in[12];
  const float* b_addr     = (const float*)d_in[13];
  const float* W_ea       = (const float*)d_in[14];
  const float* b_ea       = (const float*)d_in[15];
  float* out = (float*)d_out;

  // scratch carve-out
  char* ws = (char*)d_ws;
  size_t off = 0;
  auto carve = [&](size_t bytes) { void* pp = ws + off; off += (bytes + 255) & ~(size_t)255; return pp; };
  float*    mem_s   = (float*)carve((size_t)Bdim * Nmem * Mmem * 4);  // 256 MB working memory
  float*    gates   = (float*)carve((size_t)Bdim * G4C * 4);
  float*    c_state = (float*)carve((size_t)Bdim * Cdim * 4);
  float*    h_state = (float*)carve((size_t)Bdim * Cdim * 4);
  float*    proj    = (float*)carve((size_t)Bdim * NPROJ * 4);
  float*    biasp   = (float*)carve((size_t)NPROJ * 4);
  float*    reads   = (float*)carve((size_t)Hh * Bdim * Mmem * 4);
  _Float16* xh16    = (_Float16*)carve((size_t)Bdim * KXH * 2);
  _Float16* W116    = (_Float16*)carve((size_t)G4C * KXH * 2);
  _Float16* c16     = (_Float16*)carve((size_t)Bdim * Cdim * 2);
  _Float16* Wp16    = (_Float16*)carve((size_t)NPROJ * Cdim * 2);
  _Float16* st16    = (_Float16*)carve((size_t)Bdim * KST * 2);
  _Float16* Wo16    = (_Float16*)carve((size_t)OUTd * KST * 2);

  // controller: gates = [x_cat | h0] @ [W_ih | W_hh]^T  -> LSTM
  ntm_pack_xh<<<(Bdim*KXH + 255)/256, 256, 0, stream>>>(in_data, prev_reads, h0, xh16);
  ntm_pack_w1<<<(G4C*KXH + 255)/256, 256, 0, stream>>>(W_ih, W_hh, W116);
  ntm_gemm16<<<dim3(Bdim/16, G4C/16), 32, 0, stream>>>(xh16, W116, gates, nullptr,
                                                       Bdim, G4C, KXH, 0);
  ntm_lstm<<<(Bdim*Cdim + 255)/256, 256, 0, stream>>>(gates, b_ih, b_hh, c0, c_state, h_state);

  // all head projections in one GEMM: proj = c @ [W_addr ; W_ea]^T + bias
  ntm_pack_half<<<(Bdim*Cdim + 255)/256, 256, 0, stream>>>(c_state, c16, Bdim*Cdim);
  ntm_pack_wp<<<(NPROJ*Cdim + 255)/256, 256, 0, stream>>>(W_addr, W_ea, Wp16);
  ntm_pack_biasp<<<(NPROJ + 255)/256, 256, 0, stream>>>(b_addr, b_ea, biasp);
  ntm_gemm16<<<dim3(Bdim/16, NPROJ/16), 32, 0, stream>>>(c16, Wp16, proj, biasp,
                                                         Bdim, NPROJ, Cdim, 1);

  // sequential heads; first write head materializes the scratch memory copy
  for (int hi = 0; hi < 2 * Hh; ++hi) {
    const float* min_ = (hi <= 1) ? memory : mem_s;
    ntm_head<<<Bdim, 256, 0, stream>>>(hi, min_, mem_s, proj, prev_w, reads);
  }

  // output layer: sigmoid([h | reads] @ W_out^T + b_out)
  ntm_pack_state<<<(Bdim*KST + 255)/256, 256, 0, stream>>>(h_state, reads, st16);
  ntm_pack_half<<<(OUTd*KST + 255)/256, 256, 0, stream>>>(W_out, Wo16, OUTd*KST);
  ntm_gemm16<<<dim3(Bdim/16, OUTd/16), 32, 0, stream>>>(st16, Wo16, out, b_out,
                                                        Bdim, OUTd, KST, 2);
}
